// MsaHmmCell_17386027614771
// MI455X (gfx1250) — matrix operand order repaired and checked by reading, not compiled, not run on hardware
//
#include <hip/hip_runtime.h>

// Problem constants (match reference)
#define QQ     515      // num states
#define QP     544      // padded Q: 17 K-tiles of 32 == 34 N-tiles of 16
#define NTILES 34       // N tiles of 16
#define KTILES 17       // K tiles of 32
#define TT     512      // timesteps
#define BB     256      // batch
#define EPSV   1e-16f

typedef __attribute__((ext_vector_type(16))) _Float16 v16h;
typedef __attribute__((ext_vector_type(8)))  _Float16 v8h;
typedef __attribute__((ext_vector_type(8)))  float    v8f;

#if defined(__gfx1250__) && __has_builtin(__builtin_amdgcn_global_load_async_to_lds_b128)
#define USE_ASYNC_LDS 1
// Builtin prototype (from hipcc diagnostic): param0 is
// 'int __vector_size__(16) __device__ *'  i.e. AS1 int4*; LDS side is AS3.
typedef int v4i_vs __attribute__((vector_size(16)));
typedef __attribute__((address_space(1))) v4i_vs* gptr128;
typedef __attribute__((address_space(3))) v4i_vs* lptr128;
#endif

// ---------------------------------------------------------------------------
// Prep 1: per-row softmax stats (max, sum(exp)) for log_A rows 0..514 and
// init_logits as virtual row 515. One wave32 per row.
// ---------------------------------------------------------------------------
__global__ __launch_bounds__(256) void rowstats_kernel(
    const float* __restrict__ log_A, const float* __restrict__ init_logits,
    float* __restrict__ stats /* 516*2 floats */) {
  int row  = blockIdx.x * 8 + (threadIdx.x >> 5);
  int lane = threadIdx.x & 31;
  if (row > QQ) return;  // rows 0..515
  const float* src = (row < QQ) ? (log_A + (size_t)row * QQ) : init_logits;
  float mx = -3.4e38f;
  for (int q = lane; q < QQ; q += 32) mx = fmaxf(mx, src[q]);
  for (int off = 16; off; off >>= 1) mx = fmaxf(mx, __shfl_xor(mx, off, 32));
  float s = 0.f;
  for (int q = lane; q < QQ; q += 32) s += expf(src[q] - mx);
  for (int off = 16; off; off >>= 1) s += __shfl_xor(s, off, 32);
  if (lane == 0) { stats[row * 2] = mx; stats[row * 2 + 1] = s; }
}

// ---------------------------------------------------------------------------
// Prep 2: pack A = softmax(log_A) into f16 WMMA B-operand tiles.
// Tile (nt,kt) = 32x16 (KxN), stored (nt,kt)-major so the GEMM K-loop walks
// a constant 1024B stride (immediate offsets off one base pointer).
// Lane l: column n = nt*16 + (l&15), K = kt*32 + (l>>4)*16 + e, e=0..15
// contiguous halves per lane (32B per lane per tile).
// ---------------------------------------------------------------------------
__global__ __launch_bounds__(256) void pack_kernel(
    const float* __restrict__ log_A, const float* __restrict__ stats,
    _Float16* __restrict__ packedB) {
  int gid = blockIdx.x * 256 + threadIdx.x;
  if (gid >= NTILES * KTILES * 32) return;
  int l    = gid & 31;
  int tile = gid >> 5;
  int kt   = tile % KTILES;
  int nt   = tile / KTILES;
  int n     = nt * 16 + (l & 15);
  int kbase = kt * 32 + (l >> 4) * 16;
  v16h v;
#pragma unroll
  for (int e = 0; e < 16; ++e) {
    int k = kbase + e;
    float val = 0.f;
    if (k < QQ && n < QQ)
      val = expf(log_A[(size_t)k * QQ + n] - stats[k * 2]) / stats[k * 2 + 1];
    v[e] = (_Float16)val;
  }
  *(v16h*)(packedB + (size_t)gid * 16) = v;
}

// ---------------------------------------------------------------------------
// Prep 3: initial distribution (padded to QP, zeros beyond QQ)
// ---------------------------------------------------------------------------
__global__ __launch_bounds__(256) void initdist_kernel(
    const float* __restrict__ init_logits, const float* __restrict__ stats,
    float* __restrict__ initd) {
  int q = blockIdx.x * 256 + threadIdx.x;
  if (q >= QP) return;
  float v = 0.f;
  if (q < QQ) v = expf(init_logits[q] - stats[QQ * 2]) / stats[QQ * 2 + 1];
  initd[q] = v;
}

// ---------------------------------------------------------------------------
// Main persistent scan kernel. One WG per 16 batch rows (16 WGs total),
// 8 wave32s per WG. Forward state lives in LDS for all 512 steps.
// Emission slice for step t+1 is DMA'd into LDS asynchronously
// (GLOBAL_LOAD_ASYNC_TO_LDS_B128) overlapping the reduce/normalize phases.
// ---------------------------------------------------------------------------
__global__ __launch_bounds__(256) void hmm_forward_kernel(
    const float*   __restrict__ E,        // (T, B, QQ)
    const _Float16* __restrict__ packedB, // WMMA-B tiled A matrix (nt,kt major)
    const float*   __restrict__ initd,    // (QP)
    float*         __restrict__ out) {    // (B, T, QQ+1)
  __shared__ _Float16 fwd_h[16][QP];              // f16 forward state (A src)
  __shared__ float    anew[16][QP];               // unnormalized next forward
  __shared__ __align__(16) float ebuf[16 * QQ];   // staged emission slice
  __shared__ float    Srow[16];
  __shared__ float    llrow[16];

  const int b0   = blockIdx.x * 16;
  const int tid  = threadIdx.x;
  const int wave = tid >> 5;
  const int lane = tid & 31;

  if (tid < 16) llrow[tid] = 0.f;

  // t = 0: anew = max(E0,eps) * max(init_dist,eps)  (no transition)
  for (int m = 0; m < 16; ++m)
    for (int q = tid; q < QP; q += 256) {
      float v = 0.f;
      if (q < QQ)
        v = fmaxf(E[(size_t)(b0 + m) * QQ + q], EPSV) * fmaxf(initd[q], EPSV);
      anew[m][q] = v;
    }
  __syncthreads();

  for (int t = 0; t < TT; ++t) {
    // ---- kick off async DMA of next emission slice into LDS ----
    if (t + 1 < TT) {
      const char* gsrc =
          (const char*)(E + (size_t)(t + 1) * BB * QQ + (size_t)b0 * QQ);
      char* ldst = (char*)ebuf;
      // 16*515 floats = 2060 contiguous 16B chunks, 16B-aligned base
#if defined(USE_ASYNC_LDS)
      for (int idx = tid; idx < (16 * QQ) / 4; idx += 256)
        __builtin_amdgcn_global_load_async_to_lds_b128(
            (gptr128)(gsrc + (size_t)idx * 16),
            (lptr128)(ldst + (size_t)idx * 16), 0, 0);
#else
      for (int idx = tid; idx < (16 * QQ) / 4; idx += 256)
        ((float4*)ldst)[idx] = ((const float4*)gsrc)[idx];
#endif
    }

    // ---- row-sum reduction: wave w owns rows 2w, 2w+1 ----
#pragma unroll
    for (int rr = 0; rr < 2; ++rr) {
      int r = 2 * wave + rr;
      float s = 0.f;
      for (int q = lane; q < QP; q += 32) s += anew[r][q];
      for (int off = 16; off; off >>= 1) s += __shfl_xor(s, off, 32);
      if (lane == 0) {
        Srow[r] = s;
        llrow[r] += logf(s);
      }
    }
    __syncthreads();

    // ---- normalize, emit log(a) + ll, refresh f16 forward state ----
    for (int m = 0; m < 16; ++m) {
      float inv = 1.0f / Srow[m];
      size_t obase = (size_t)(b0 + m) * TT * (QQ + 1) + (size_t)t * (QQ + 1);
      for (int q = tid; q < QP; q += 256) {
        float a = anew[m][q] * inv;
        fwd_h[m][q] = (_Float16)a;
        if (q < QQ) out[obase + q] = logf(a);
      }
    }
    if (tid < 16)
      out[(size_t)(b0 + tid) * TT * (QQ + 1) + (size_t)t * (QQ + 1) + QQ] =
          llrow[tid];

    // wait for this wave's async DMA, then barrier so all LDS writes visible
#if defined(USE_ASYNC_LDS)
#if __has_builtin(__builtin_amdgcn_s_wait_asynccnt)
    __builtin_amdgcn_s_wait_asynccnt(0);
#else
    asm volatile("s_wait_asynccnt 0x0" ::: "memory");
#endif
#endif
    __syncthreads();

    if (t + 1 >= TT) break;

    // ---- load A fragments once per step (wave-invariant across nt) ----
    const int am = lane & 15;  // A-matrix row (batch)
    const int kh = lane >> 4;  // K half select
    v16h afrag[KTILES];
#pragma unroll
    for (int kt = 0; kt < KTILES; ++kt) {
      // 16-bit A layout: halves 0-15 = K kt*32+kh*16.. (contiguous, matches
      // packed B)  -> two 16B LDS vector loads
      v8h lo = *(const v8h*)&fwd_h[am][kt * 32 + kh * 16];
      v8h hi = *(const v8h*)&fwd_h[am][kt * 32 + kh * 16 + 8];
      afrag[kt] = __builtin_shufflevector(
          lo, hi, 0, 1, 2, 3, 4, 5, 6, 7, 8, 9, 10, 11, 12, 13, 14, 15);
    }

    // ---- R = fwd @ A via f16 WMMA, fused emission epilogue -> anew ----
    for (int nt = wave; nt < NTILES; nt += 8) {
      const _Float16* bp = packedB + ((size_t)nt * KTILES * 32 + lane) * 16;
      v8f acc = {0.f, 0.f, 0.f, 0.f, 0.f, 0.f, 0.f, 0.f};
#pragma unroll
      for (int kt = 0; kt < KTILES; ++kt) {
        v16h bfrag = *(const v16h*)(bp + (size_t)kt * 512);  // +1024B/kt
        acc = __builtin_amdgcn_wmma_f32_16x16x32_f16(
            false, afrag[kt], false, bfrag, (short)0, acc, false, false);
      }
      // C/D layout: lanes 0-15 -> M = v, lanes 16-31 -> M = v+8; N = lane&15
      const int n = lane & 15;
      const int q = nt * 16 + n;
#pragma unroll
      for (int v = 0; v < 8; ++v) {
        int m = (lane < 16) ? v : (v + 8);
        float val = 0.f;
        if (q < QQ)
          val = fmaxf(ebuf[m * QQ + q], EPSV) * fmaxf(acc[v], EPSV);
        anew[m][q] = val;
      }
    }
    __syncthreads();
  }
}

// ---------------------------------------------------------------------------
extern "C" void kernel_launch(void* const* d_in, const int* in_sizes, int n_in,
                              void* d_out, int out_size, void* d_ws,
                              size_t ws_size, hipStream_t stream) {
  const float* E           = (const float*)d_in[0];  // (T, 1, B, Q)
  const float* log_A       = (const float*)d_in[1];  // (1, Q, Q)
  const float* init_logits = (const float*)d_in[2];  // (1, Q)
  float* out = (float*)d_out;                        // (B, T, Q+1)

  char* ws = (char*)d_ws;
  float*    stats   = (float*)ws;                       // 516*2 f32
  _Float16* packedB = (_Float16*)(ws + 8192);           // 34*17*32*16 halves
  float*    initd   = (float*)(ws + 8192 + 591872);     // QP f32

  rowstats_kernel<<<(QQ + 1 + 7) / 8, 256, 0, stream>>>(log_A, init_logits,
                                                        stats);
  pack_kernel<<<(NTILES * KTILES * 32 + 255) / 256, 256, 0, stream>>>(
      log_A, stats, packedB);
  initdist_kernel<<<(QP + 255) / 256, 256, 0, stream>>>(init_logits, stats,
                                                        initd);
  hmm_forward_kernel<<<BB / 16, 256, 0, stream>>>(E, packedB, initd, out);
}
